// AttentionHead_17248588661523
// MI455X (gfx1250) — matrix operand order repaired and checked by reading, not compiled, not run on hardware
//
#include <hip/hip_runtime.h>
#include <hip/hip_bf16.h>

// CDNA5 / gfx1250, wave32 only.
typedef __attribute__((ext_vector_type(16))) __bf16 v16bf;
typedef __attribute__((ext_vector_type(8)))  __bf16 v8bf;
typedef __attribute__((ext_vector_type(8)))  float  v8f;
typedef __attribute__((ext_vector_type(4)))  float  v4f;

#define N_EMB   1024
#define N_HEADS 16
#define HS      64
#define BT      4096   // B*T
#define TSEQ    2048

static __device__ __forceinline__ v8f wmma_bf16(v16bf a, v16bf b, v8f c) {
  // D = A(16x32 bf16) * B(32x16 bf16) + C(16x16 f32)
  return __builtin_amdgcn_wmma_f32_16x16x32_bf16(false, a, false, b,
                                                 (short)0, c, false, false);
}

static __device__ __forceinline__ v16bf cat8(v8bf a, v8bf b) {
  return __builtin_shufflevector(a, b, 0, 1, 2, 3, 4, 5, 6, 7,
                                       8, 9, 10, 11, 12, 13, 14, 15);
}

// ---------------------------------------------------------------------------
// Kernel 1: Q/K/V = x @ W^T (bf16 WMMA, f32 accum), RoPE on Q,K.
// One wave computes one 16x16 output tile. All fragment loads are 16B vector
// loads (global_load_b128) feeding v_cvt_pk_bf16_f32.
//   Q,K: bf16 [b][h][T][64] row-major.  V: bf16 [b][h][64][T] (transposed).
// ---------------------------------------------------------------------------
__global__ void qkv_rope_wmma(const float* __restrict__ x,
                              const float* __restrict__ Wq,
                              const float* __restrict__ Wk,
                              const float* __restrict__ Wv,
                              __bf16* __restrict__ Q,
                              __bf16* __restrict__ K,
                              __bf16* __restrict__ Vt) {
  const int lane = threadIdx.x;
  const int lo = lane & 15, hi = lane >> 4;
  const int r0 = blockIdx.x * 16;     // row tile in BT
  const int o0 = blockIdx.y * 16;     // output-feature tile in C
  const int mat = blockIdx.z;         // 0=q 1=k 2=v
  const float* W = (mat == 0) ? Wq : ((mat == 1) ? Wk : Wv);

  v8f acc = {};   // C/D layout: lane=col n=o0+lo, rows m = v + 8*hi

  for (int c0 = 0; c0 < N_EMB; c0 += 32) {
    // A fragment: x rows. lane row = r0+lo; elems {0-15} / {16-31}
    const float* xrow = x + (size_t)(r0 + lo) * N_EMB + c0;
    const v4f* xa = (const v4f*)(xrow + 16 * hi);        // 32B-aligned
    const v4f* xb = (const v4f*)(xrow + 16 * hi + 8);
    const v4f f0 = xa[0], f1 = xa[1], f2 = xb[0], f3 = xb[1];
    v16bf a;
#pragma unroll
    for (int j = 0; j < 4; ++j) {
      a[j]      = (__bf16)f0[j];
      a[4 + j]  = (__bf16)f1[j];
      a[8 + j]  = (__bf16)f2[j];
      a[12 + j] = (__bf16)f3[j];
    }
    // B fragment: B[k][n] = W[o0+n][c0+k]; lane col = o0+lo; k = j + 16*hi
    const v4f* wp = (const v4f*)(W + (size_t)(o0 + lo) * N_EMB + c0 + 16 * hi);
    const v4f w0 = wp[0], w1 = wp[1], w2 = wp[2], w3 = wp[3];
    v16bf bf;
#pragma unroll
    for (int j = 0; j < 4; ++j) {
      bf[j]      = (__bf16)w0[j];
      bf[4 + j]  = (__bf16)w1[j];
      bf[8 + j]  = (__bf16)w2[j];
      bf[12 + j] = (__bf16)w3[j];
    }
    acc = wmma_bf16(a, bf, acc);
  }

  const int b  = r0 >> 11;          // T = 2048
  const int t0 = r0 & (TSEQ - 1);
  const int o  = o0 + lo;           // global output feature
  const int h  = o >> 6;
  const int d  = o & 63;

  if (mat < 2) {
    // RoPE: pairs (2i,2i+1) are adjacent lanes in the C/D layout.
    const int i = d >> 1;
    // theta = 10000^(-2i/1024) = exp(-i * 2*ln(10000)/1024)
    const float theta = __expf(-(float)i * (18.420680743952367f / 1024.0f));
    const bool even = (d & 1) == 0;
#pragma unroll
    for (int v = 0; v < 8; ++v) {
      const float t = (float)(t0 + v + 8 * hi);
      const float partner = __shfl_xor(acc[v], 1, 32);
      float s, c;
      __sincosf(t * theta, &s, &c);
      acc[v] = even ? (acc[v] * c - partner * s)
                    : (acc[v] * c + partner * s);
    }
    __bf16* dst = (mat == 0 ? Q : K) +
                  (((size_t)(b * N_HEADS + h) * TSEQ + t0) * HS + d);
#pragma unroll
    for (int v = 0; v < 8; ++v)
      dst[(size_t)(v + 8 * hi) * HS] = (__bf16)acc[v];
  } else {
    // V transposed: [b][h][d][t] -> one 16B packed store per lane
    v8bf pk;
#pragma unroll
    for (int v = 0; v < 8; ++v) pk[v] = (__bf16)acc[v];
    __bf16* dst = Vt + (((size_t)(b * N_HEADS + h) * HS + d) * TSEQ + t0 + 8 * hi);
    *(v8bf*)dst = pk;
  }
}

// ---------------------------------------------------------------------------
// Kernel 2: causal flash attention. One wave owns one 16-row q-tile of one
// (b,h). K consumed in 32-wide blocks: S = 4 WMMAs, online softmax with
// half-wave butterfly row reductions, P transposed via LDS, O += P*V (4 WMMAs).
// All fragment loads are aligned 16B vector loads; next K block prefetched.
// ---------------------------------------------------------------------------
__global__ void flash_attn_wmma(const __bf16* __restrict__ Q,
                                const __bf16* __restrict__ K,
                                const __bf16* __restrict__ Vt,
                                float* __restrict__ out) {
  __shared__ __bf16 lds_p[16][48];   // 16(q) x 32(k), row stride 96B (16B mult)

  const int lane = threadIdx.x;
  const int lo = lane & 15, hi = lane >> 4;
  const int q0 = blockIdx.x * 16;
  const int h  = blockIdx.y;
  const int b  = blockIdx.z;

  const size_t bh = (size_t)(b * N_HEADS + h);
  const __bf16* Qp = Q  + (bh * TSEQ + q0) * HS;
  const __bf16* Kp = K  + bh * TSEQ * HS;
  const __bf16* Vp = Vt + bh * HS * TSEQ;

  // Q A-fragments (two 32-wide d chunks); softmax scale folded into S later.
  v16bf aq0, aq1;
  {
    const __bf16* qrow = Qp + (size_t)lo * HS;
    const v8bf qa = *(const v8bf*)(qrow + 16 * hi);
    const v8bf qb = *(const v8bf*)(qrow + 8 + 16 * hi);
    const v8bf qc = *(const v8bf*)(qrow + 32 + 16 * hi);
    const v8bf qd = *(const v8bf*)(qrow + 40 + 16 * hi);
    aq0 = cat8(qa, qb);
    aq1 = cat8(qc, qd);
  }

  float mrow[8], lrow[8];
  v8f oacc[4] = {{}, {}, {}, {}};
#pragma unroll
  for (int v = 0; v < 8; ++v) { mrow[v] = -__builtin_inff(); lrow[v] = 0.0f; }

  const int nkb = (q0 + 16 + 31) >> 5;     // causal: k <= q0+15
  for (int kb = 0; kb < nkb; ++kb) {
    const int k0 = kb * 32;

    if (kb + 1 < nkb) {  // gfx1250 global_prefetch_b8 for the next K block
      __builtin_prefetch(Kp + (size_t)(k0 + 32 + lo) * HS, 0, 1);
      __builtin_prefetch(Kp + (size_t)(k0 + 48 + lo) * HS, 0, 1);
    }

    // ---- S = Q K^T : two 16x16 tiles, each 2 WMMAs over d ----
    v8f s0 = {}, s1 = {};
    {
      const __bf16* kr0 = Kp + (size_t)(k0 + lo) * HS + 16 * hi;
      v16bf bk = cat8(*(const v8bf*)kr0, *(const v8bf*)(kr0 + 8));
      s0 = wmma_bf16(aq0, bk, s0);
      bk = cat8(*(const v8bf*)(kr0 + 32), *(const v8bf*)(kr0 + 40));
      s0 = wmma_bf16(aq1, bk, s0);

      const __bf16* kr1 = Kp + (size_t)(k0 + 16 + lo) * HS + 16 * hi;
      bk = cat8(*(const v8bf*)kr1, *(const v8bf*)(kr1 + 8));
      s1 = wmma_bf16(aq0, bk, s1);
      bk = cat8(*(const v8bf*)(kr1 + 32), *(const v8bf*)(kr1 + 40));
      s1 = wmma_bf16(aq1, bk, s1);
    }

    // ---- causal mask + C^-0.5 scale + online softmax ----
#pragma unroll
    for (int v = 0; v < 8; ++v) {
      const int qr = q0 + v + 8 * hi;
      const float a0 = (k0 + lo      <= qr) ? s0[v] * 0.03125f : -__builtin_inff();
      const float a1 = (k0 + 16 + lo <= qr) ? s1[v] * 0.03125f : -__builtin_inff();
      float bm = fmaxf(a0, a1);
      bm = fmaxf(bm, __shfl_xor(bm, 1, 32));
      bm = fmaxf(bm, __shfl_xor(bm, 2, 32));
      bm = fmaxf(bm, __shfl_xor(bm, 4, 32));
      bm = fmaxf(bm, __shfl_xor(bm, 8, 32));
      const float newm  = fmaxf(mrow[v], bm);
      const float alpha = __expf(mrow[v] - newm);
      const float p0 = __expf(a0 - newm);
      const float p1 = __expf(a1 - newm);
      float bs = p0 + p1;
      bs += __shfl_xor(bs, 1, 32);
      bs += __shfl_xor(bs, 2, 32);
      bs += __shfl_xor(bs, 4, 32);
      bs += __shfl_xor(bs, 8, 32);
      lrow[v] = lrow[v] * alpha + bs;
      mrow[v] = newm;
      oacc[0][v] *= alpha; oacc[1][v] *= alpha;
      oacc[2][v] *= alpha; oacc[3][v] *= alpha;
      lds_p[v + 8 * hi][lo]      = (__bf16)p0;   // P in (q,k) order
      lds_p[v + 8 * hi][16 + lo] = (__bf16)p1;
    }
    __syncthreads();

    // ---- P A-fragment from LDS (C/D layout -> A layout), ds_load_b128 ----
    const v16bf ap = cat8(*(const v8bf*)&lds_p[lo][16 * hi],
                          *(const v8bf*)&lds_p[lo][8 + 16 * hi]);

    // ---- O += P * V  (V transposed -> contiguous 16B B-fragment loads) ----
#pragma unroll
    for (int dt = 0; dt < 4; ++dt) {
      const __bf16* vrow = Vp + (size_t)(dt * 16 + lo) * TSEQ + k0 + 16 * hi;
      const v16bf bv = cat8(*(const v8bf*)vrow, *(const v8bf*)(vrow + 8));
      oacc[dt] = wmma_bf16(ap, bv, oacc[dt]);
    }
    __syncthreads();
  }

  // ---- normalize and store fp32 output (B,T,C) ----
  float* op = out + ((size_t)b * TSEQ + q0) * N_EMB + h * HS;
#pragma unroll
  for (int dt = 0; dt < 4; ++dt)
#pragma unroll
    for (int v = 0; v < 8; ++v)
      op[(size_t)(v + 8 * hi) * N_EMB + dt * 16 + lo] = oacc[dt][v] / lrow[v];
}

// ---------------------------------------------------------------------------
extern "C" void kernel_launch(void* const* d_in, const int* in_sizes, int n_in,
                              void* d_out, int out_size, void* d_ws, size_t ws_size,
                              hipStream_t stream) {
  const float* x  = (const float*)d_in[0];
  const float* Wq = (const float*)d_in[1];
  const float* Wk = (const float*)d_in[2];
  const float* Wv = (const float*)d_in[3];
  float* out = (float*)d_out;

  // Workspace: Q,K row-major bf16 + V transposed bf16 = 3 * 8 MB = 24 MB
  const size_t qkv_elems = (size_t)2 * N_HEADS * TSEQ * HS;  // 4 Mi elements
  __bf16* Q  = (__bf16*)d_ws;
  __bf16* K  = Q + qkv_elems;
  __bf16* Vt = K + qkv_elems;

  qkv_rope_wmma<<<dim3(BT / 16, N_EMB / 16, 3), 32, 0, stream>>>(
      x, Wq, Wk, Wv, Q, K, Vt);
  flash_attn_wmma<<<dim3(TSEQ / 16, N_HEADS, 2), 32, 0, stream>>>(
      Q, K, Vt, out);
}
